// CustomMultiheadAttention_80900003987708
// MI455X (gfx1250) — compile-verified, matrix operand order checked by reading, not run on hardware
//
#include <hip/hip_runtime.h>
#include <hip/hip_bf16.h>
#include <math.h>

// ---------------------------------------------------------------------------
// Problem dims (fixed by the reference)
// ---------------------------------------------------------------------------
#define B_    2
#define SQ_   2048
#define SKV_  2048
#define DQ_   1024
#define DKV_  1024
#define H_    1024
#define NH_   16
#define HD_   64          // head dim
#define ROWS_ (B_ * SQ_)  // 4096 token rows

typedef __attribute__((ext_vector_type(16))) _Float16 v16h;
typedef __attribute__((ext_vector_type(8)))  _Float16 v8h;
typedef __attribute__((ext_vector_type(8)))  float    v8f;
typedef __attribute__((ext_vector_type(4)))  int      v4i;

// Pointer types the async-copy builtin expects (probed from the round-2
// diagnostic: param0 is "int4 addrspace(1)*", i.e. global src first).
typedef __attribute__((address_space(1))) v4i* gv4i_p;
typedef __attribute__((address_space(3))) v4i* lv4i_p;

// ---------------------------------------------------------------------------
// Async global->LDS copy (CDNA5 GLOBAL_LOAD_ASYNC_TO_LDS_B128, ASYNCcnt).
// Guarded: falls back to a synchronous b128 copy if the builtin is absent.
// ---------------------------------------------------------------------------
#if defined(__has_builtin)
#if __has_builtin(__builtin_amdgcn_global_load_async_to_lds_b128)
#define HAVE_ASYNC_G2L 1
#endif
#endif
#ifndef HAVE_ASYNC_G2L
#define HAVE_ASYNC_G2L 0
#endif

#if HAVE_ASYNC_G2L
#if __has_builtin(__builtin_amdgcn_s_wait_asynccnt)
#define WAIT_ASYNC(n) __builtin_amdgcn_s_wait_asynccnt(n)
#else
#define WAIT_ASYNC(n) asm volatile("s_wait_asynccnt " #n ::: "memory")
#endif
#else
#define WAIT_ASYNC(n)
#endif

__device__ __forceinline__ void g2l_b128(const _Float16* g, _Float16* l) {
#if HAVE_ASYNC_G2L
  __builtin_amdgcn_global_load_async_to_lds_b128((gv4i_p)g, (lv4i_p)l, 0, 0);
#else
  *(v8h*)__builtin_assume_aligned(l, 16) =
      *(const v8h*)__builtin_assume_aligned(g, 16);
#endif
}

// ---------------------------------------------------------------------------
// WMMA helper: D = A(16x32 f16) * B(32x16 f16) + C(16x16 f32)
// ---------------------------------------------------------------------------
__device__ __forceinline__ v8f wmma_f16(v16h a, v16h b, v8f c) {
  return __builtin_amdgcn_wmma_f32_16x16x32_f16(
      /*neg_a=*/false, a, /*neg_b=*/false, b,
      /*c_mod=*/(short)0, c, /*reuse_a=*/false, /*reuse_b=*/false);
}

// ---------------------------------------------------------------------------
// Fragment loaders.
// Per ISA 7.12.2 (16-bit A-matrix 16x32, wave32):
//   row (A) / col (B)           = lane & 15
//   element h of the v16h frag  -> k = (h&8)*2 + ((lane>>4)<<3) + (h&7)
// i.e. two contiguous 8-half (16 B) runs at k_base and k_base+16.
// B-fragments use the same loader on a transposed (row = n, col = k) matrix.
// All call sites keep 16-byte alignment (ld and k0 are multiples of 8 halves).
// ---------------------------------------------------------------------------
__device__ __forceinline__ v16h load_frag_f16(const _Float16* p, int ld,
                                              int row0, int k0) {
  int lane = threadIdx.x & 31;
  const _Float16* base = (const _Float16*)__builtin_assume_aligned(
      p + (size_t)(row0 + (lane & 15)) * ld + k0 + ((lane >> 4) << 3), 16);
  v16h f;
#pragma unroll
  for (int i = 0; i < 8; ++i) { f[i] = base[i]; f[i + 8] = base[16 + i]; }
  return f;
}

__device__ __forceinline__ v16h load_frag_f32(const float* p, int ld,
                                              int row0, int k0) {
  int lane = threadIdx.x & 31;
  const float* base = (const float*)__builtin_assume_aligned(
      p + (size_t)(row0 + (lane & 15)) * ld + k0 + ((lane >> 4) << 3), 16);
  v16h f;
#pragma unroll
  for (int i = 0; i < 8; ++i) {
    f[i]     = (_Float16)base[i];
    f[i + 8] = (_Float16)base[16 + i];
  }
  return f;
}

// A-frag loader over the head-interleaved O buffer (b, h, q, hd).
// Each 8-element k-run (8-aligned) stays inside one head because HD_ = 64.
__device__ __forceinline__ v16h load_frag_ohead(const _Float16* o_h,
                                                int row0, int k0) {
  int lane = threadIdx.x & 31;
  int rm = row0 + (lane & 15);
  int bb = rm >> 11;              // / SQ_
  int q  = rm & (SQ_ - 1);
  int kbase = k0 + ((lane >> 4) << 3);
  v16h f;
#pragma unroll
  for (int t = 0; t < 2; ++t) {
    int kk = kbase + t * 16;
    int h  = kk >> 6;
    int hd = kk & 63;
    const _Float16* base = (const _Float16*)__builtin_assume_aligned(
        o_h + ((size_t)((bb * NH_ + h) * SQ_ + q)) * HD_ + hd, 16);
#pragma unroll
    for (int i = 0; i < 8; ++i) f[t * 8 + i] = base[i];
  }
  return f;
}

// ---------------------------------------------------------------------------
// Kernel 0: weight transpose + f32 -> f16  (wt[n*K + k] = w[k*N + n])
// ---------------------------------------------------------------------------
__global__ void wt_f16_kernel(const float* __restrict__ w,
                              _Float16* __restrict__ wt, int K, int N) {
  int idx = blockIdx.x * blockDim.x + threadIdx.x;
  if (idx >= K * N) return;
  int k = idx / N, n = idx - k * N;
  wt[(size_t)n * K + k] = (_Float16)w[idx];
}

// ---------------------------------------------------------------------------
// Kernel 1: fused Q/K/V projection (blockIdx.z selects which).
// Each wave: 16x64 output tile, K-loop step 32, 4 WMMAs / step.
//   which 0 -> Q  stored (b, h, sq, 64)   f16
//   which 1 -> K  stored (b, h, skv, 64)  f16   (== Bt layout for Q*K^T)
//   which 2 -> V  stored (b, h, 64, skv)  f16   (== Bt layout for P*V)
// ---------------------------------------------------------------------------
__global__ void __launch_bounds__(256) proj_kernel(
    const float* __restrict__ xq, const float* __restrict__ xkv,
    const _Float16* __restrict__ wqt, const _Float16* __restrict__ wkt,
    const _Float16* __restrict__ wvt,
    const float* __restrict__ bq, const float* __restrict__ bk,
    const float* __restrict__ bv,
    _Float16* __restrict__ q_h, _Float16* __restrict__ k_h,
    _Float16* __restrict__ v_t) {
  int which = blockIdx.z;
  const float*    x    = (which == 0) ? xq  : xkv;
  const _Float16* wt   = (which == 0) ? wqt : (which == 1) ? wkt : wvt;
  const float*    bias = (which == 0) ? bq  : (which == 1) ? bk  : bv;

  int wave = threadIdx.x >> 5;
  int lane = threadIdx.x & 31;
  int row0 = blockIdx.x * 128 + wave * 16;  // token row
  int n0   = blockIdx.y * 64;               // output feature

  v8f acc[4] = {};
  for (int k0 = 0; k0 < DQ_; k0 += 32) {
    v16h a = load_frag_f32(x, DQ_, row0, k0);
#pragma unroll
    for (int j = 0; j < 4; ++j) {
      v16h b = load_frag_f16(wt, DQ_, n0 + 16 * j, k0);
      acc[j] = wmma_f16(a, b, acc[j]);
    }
  }

  int half = lane >> 4, col = lane & 15;
#pragma unroll
  for (int j = 0; j < 4; ++j) {
    int n = n0 + 16 * j + col;
    int h = n >> 6, hd = n & 63;
    float bias_n = bias[n];
#pragma unroll
    for (int r = 0; r < 8; ++r) {
      int rm = row0 + r + half * 8;
      int bb = rm >> 11;          // / SQ_
      int s  = rm & (SQ_ - 1);
      _Float16 val = (_Float16)(acc[j][r] + bias_n);
      if (which == 0)
        q_h[((size_t)((bb * NH_ + h) * SQ_ + s)) * HD_ + hd] = val;
      else if (which == 1)
        k_h[((size_t)((bb * NH_ + h) * SKV_ + s)) * HD_ + hd] = val;
      else
        v_t[((size_t)((bb * NH_ + h) * HD_ + hd)) * SKV_ + s] = val;
    }
  }
}

// ---------------------------------------------------------------------------
// Kernel 2: flash attention with double-buffered async K/V staging.
// Block = 8 waves, one (b,h), 128 queries. Per 32-key step the block stages
// K (32x64) and V^T (64x32) tiles into LDS (async global->LDS, ASYNCcnt),
// overlapping the next tile's DMA with the current tile's WMMAs.
// Online softmax row stats via shfl_xor over 16-lane groups; P transposed
// C-layout -> A-layout through a 1 KB/wave LDS tile (s_wait_dscnt guarded).
// ---------------------------------------------------------------------------
#define MASK_NEG (-1e30f)

__device__ __forceinline__ void stage_tiles(const _Float16* kp,
                                            const _Float16* vp, int kb,
                                            _Float16* kt, _Float16* vt,
                                            int tid) {
  // K tile: 32 rows x 64 halves, one b128 per thread
  int kr = tid >> 3, kc = (tid & 7) << 3;
  g2l_b128(kp + (size_t)(kb + kr) * HD_ + kc, kt + kr * HD_ + kc);
  // V^T tile: 64 rows x 32 halves, one b128 per thread
  int vr = tid >> 2, vc = (tid & 3) << 3;
  g2l_b128(vp + (size_t)vr * SKV_ + kb + vc, vt + vr * 32 + vc);
}

__global__ void __launch_bounds__(256) flash_attn_kernel(
    const _Float16* __restrict__ q_h, const _Float16* __restrict__ k_h,
    const _Float16* __restrict__ v_t, const unsigned char* __restrict__ mask,
    _Float16* __restrict__ o_h) {
  __shared__ __align__(16) _Float16 k_tile[2][32 * 64];  // 8 KB
  __shared__ __align__(16) _Float16 v_tile[2][64 * 32];  // 8 KB
  __shared__ __align__(16) _Float16 p_lds[8][16 * 32];   // 8 KB

  int tid  = threadIdx.x;
  int wave = tid >> 5;
  int lane = tid & 31;
  int bh   = blockIdx.y;                    // b*NH + h
  int bb   = bh / NH_;
  int q0   = blockIdx.x * 128 + wave * 16;

  const _Float16* qp = q_h + (size_t)bh * SQ_ * HD_;
  const _Float16* kp = k_h + (size_t)bh * SKV_ * HD_;
  const _Float16* vp = v_t + (size_t)bh * HD_ * SKV_;
  const unsigned char* mp = mask + (size_t)bb * SKV_;

  v16h qa0 = load_frag_f16(qp, HD_, q0, 0);
  v16h qa1 = load_frag_f16(qp, HD_, q0, 32);

  v8f o[4] = {};
  float mrow[8], lrow[8];
#pragma unroll
  for (int r = 0; r < 8; ++r) { mrow[r] = MASK_NEG; lrow[r] = 0.f; }

  int half = lane >> 4, col = lane & 15;
  _Float16* lds = &p_lds[wave][0];
  const float scale = 0.125f;               // 1/sqrt(64)

  // prologue: stage tile 0
  stage_tiles(kp, vp, 0, k_tile[0], v_tile[0], tid);

  for (int kb = 0; kb < SKV_; kb += 32) {
    int cur = (kb >> 5) & 1;
    if (kb + 32 < SKV_) {
      stage_tiles(kp, vp, kb + 32, k_tile[cur ^ 1], v_tile[cur ^ 1], tid);
      WAIT_ASYNC(2);                        // current tile's 2 loads done
    } else {
      WAIT_ASYNC(0);
    }
    __syncthreads();                        // tile visible to whole block

    const _Float16* kt = k_tile[cur];
    const _Float16* vt = v_tile[cur];

    // S tiles: 16 queries x 32 keys (two 16-key WMMAs, HD contracted in 2x32)
    v8f s0 = {}, s1 = {};
    s0 = wmma_f16(qa0, load_frag_f16(kt, HD_, 0, 0), s0);
    s0 = wmma_f16(qa1, load_frag_f16(kt, HD_, 0, 32), s0);
    s1 = wmma_f16(qa0, load_frag_f16(kt, HD_, 16, 0), s1);
    s1 = wmma_f16(qa1, load_frag_f16(kt, HD_, 16, 32), s1);

    bool m0 = mp[kb + col] != 0;            // key-padding mask, per column
    bool m1 = mp[kb + 16 + col] != 0;

    float mnew[8];
#pragma unroll
    for (int r = 0; r < 8; ++r) {
      float a0 = m0 ? MASK_NEG : (float)s0[r] * scale;
      float a1 = m1 ? MASK_NEG : (float)s1[r] * scale;
      s0[r] = a0; s1[r] = a1;
      float v = fmaxf(a0, a1);
      v = fmaxf(v, __shfl_xor(v, 1));
      v = fmaxf(v, __shfl_xor(v, 2));
      v = fmaxf(v, __shfl_xor(v, 4));
      v = fmaxf(v, __shfl_xor(v, 8));       // row max within 16-lane group
      mnew[r] = fmaxf(mrow[r], v);
    }

#pragma unroll
    for (int r = 0; r < 8; ++r) {
      float c  = __expf(mrow[r] - mnew[r]);
      float p0 = __expf(s0[r] - mnew[r]);
      float p1 = __expf(s1[r] - mnew[r]);
      float ps = p0 + p1;
      ps += __shfl_xor(ps, 1);
      ps += __shfl_xor(ps, 2);
      ps += __shfl_xor(ps, 4);
      ps += __shfl_xor(ps, 8);
      lrow[r] = lrow[r] * c + ps;
      mrow[r] = mnew[r];
      o[0][r] *= c; o[1][r] *= c; o[2][r] *= c; o[3][r] *= c;
      int row = r + half * 8;               // C-layout -> (row, col) in LDS
      lds[row * 32 + col]      = (_Float16)p0;
      lds[row * 32 + 16 + col] = (_Float16)p1;
    }

    // LDS P-tile is per-wave private and DS ops are in-order per wave; the
    // explicit dscnt wait makes the store->load ordering airtight.
    asm volatile("s_wait_dscnt 0" ::: "memory");

    v16h pa = load_frag_f16(lds, 32, 0, 0); // P in A-layout
#pragma unroll
    for (int j = 0; j < 4; ++j) {
      v16h bvf = load_frag_f16(vt, 32, 16 * j, 0);  // V^T rows = hd
      o[j] = wmma_f16(pa, bvf, o[j]);
    }

    __syncthreads();  // everyone done reading tile `cur` before it's restaged
  }

  // normalize + store O as (b, h, q, hd) f16
#pragma unroll
  for (int j = 0; j < 4; ++j) {
    int hd = 16 * j + col;
#pragma unroll
    for (int r = 0; r < 8; ++r) {
      int q = q0 + r + half * 8;
      o_h[((size_t)bh * SQ_ + q) * HD_ + hd] = (_Float16)(o[j][r] / lrow[r]);
    }
  }
}

// ---------------------------------------------------------------------------
// Kernel 3: output projection  out = O @ Wo + bo   (fp32 result)
// ---------------------------------------------------------------------------
__global__ void __launch_bounds__(256) out_proj_kernel(
    const _Float16* __restrict__ o_h, const _Float16* __restrict__ wot,
    const float* __restrict__ bo, float* __restrict__ out) {
  int wave = threadIdx.x >> 5;
  int lane = threadIdx.x & 31;
  int row0 = blockIdx.x * 128 + wave * 16;
  int n0   = blockIdx.y * 64;

  v8f acc[4] = {};
  for (int k0 = 0; k0 < H_; k0 += 32) {
    v16h a = load_frag_ohead(o_h, row0, k0);
#pragma unroll
    for (int j = 0; j < 4; ++j) {
      v16h b = load_frag_f16(wot, H_, n0 + 16 * j, k0);
      acc[j] = wmma_f16(a, b, acc[j]);
    }
  }

  int half = lane >> 4, col = lane & 15;
#pragma unroll
  for (int j = 0; j < 4; ++j) {
    int n = n0 + 16 * j + col;
    float bias_n = bo[n];
#pragma unroll
    for (int r = 0; r < 8; ++r) {
      int rm = row0 + r + half * 8;
      out[(size_t)rm * DQ_ + n] = acc[j][r] + bias_n;
    }
  }
}

// ---------------------------------------------------------------------------
// Launcher. Workspace layout (40 MB total):
//   [0,8)MB  WqT/WkT/WvT/WoT f16   [8,16) Q  [16,24) K  [24,32) V^T  [32,40) O
// ---------------------------------------------------------------------------
extern "C" void kernel_launch(void* const* d_in, const int* in_sizes, int n_in,
                              void* d_out, int out_size, void* d_ws,
                              size_t ws_size, hipStream_t stream) {
  const float* xq   = (const float*)d_in[0];                 // query_input
  const float* xkv  = (const float*)d_in[1];                 // key_value_input
  const unsigned char* mask = (const unsigned char*)d_in[2]; // bool mask
  const float* Wq = (const float*)d_in[3];
  const float* bq = (const float*)d_in[4];
  const float* Wk = (const float*)d_in[5];
  const float* bk = (const float*)d_in[6];
  const float* Wv = (const float*)d_in[7];
  const float* bv = (const float*)d_in[8];
  const float* Wo = (const float*)d_in[9];
  const float* bo = (const float*)d_in[10];
  float* out = (float*)d_out;

  char* ws = (char*)d_ws;
  const size_t MB = 1024ull * 1024ull;
  _Float16* wqt = (_Float16*)(ws + 0 * MB);
  _Float16* wkt = (_Float16*)(ws + 2 * MB);
  _Float16* wvt = (_Float16*)(ws + 4 * MB);
  _Float16* wot = (_Float16*)(ws + 6 * MB);
  _Float16* q_h = (_Float16*)(ws + 8 * MB);
  _Float16* k_h = (_Float16*)(ws + 16 * MB);
  _Float16* v_t = (_Float16*)(ws + 24 * MB);
  _Float16* o_h = (_Float16*)(ws + 32 * MB);

  // 0) weight transpose + f16 convert
  {
    int n = DQ_ * H_;
    int blocks = (n + 255) / 256;
    wt_f16_kernel<<<blocks, 256, 0, stream>>>(Wq, wqt, DQ_, H_);
    wt_f16_kernel<<<blocks, 256, 0, stream>>>(Wk, wkt, DKV_, H_);
    wt_f16_kernel<<<blocks, 256, 0, stream>>>(Wv, wvt, DKV_, H_);
    wt_f16_kernel<<<blocks, 256, 0, stream>>>(Wo, wot, H_, DQ_);
  }

  // 1) fused QKV projection (z = 0/1/2 -> Q/K/V)
  proj_kernel<<<dim3(ROWS_ / 128, H_ / 64, 3), 256, 0, stream>>>(
      xq, xkv, wqt, wkt, wvt, bq, bk, bv, q_h, k_h, v_t);

  // 2) flash attention: 16 q-tiles/block-x, (b,h) on y
  flash_attn_kernel<<<dim3(SQ_ / 128, B_ * NH_), 256, 0, stream>>>(
      q_h, k_h, v_t, mask, o_h);

  // 3) output projection
  out_proj_kernel<<<dim3(ROWS_ / 128, DQ_ / 64), 256, 0, stream>>>(
      o_h, wot, bo, out);
}
